// MLMCharaterLearning_24120536334837
// MI455X (gfx1250) — compile-verified, matrix operand order checked by reading, not run on hardware
//
#include <hip/hip_runtime.h>
#include <hip/hip_bf16.h>

typedef __attribute__((ext_vector_type(16))) __bf16 v16bf;
typedef __attribute__((ext_vector_type(8)))  float  v8f;

static constexpr int HD    = 1024;    // hidden
static constexpr int VOC   = 32000;   // vocab
static constexpr int MTOK  = 8192;    // B*T tokens
static constexpr int VTILE = 128;     // vocab tile per iteration
static constexpr int NWAVE = 8;       // waves per block (256 threads, wave32)

// ---------------------------------------------------------------------------
// Kernel 1: W fp32 -> bf16 row-major copy (for GEMM1 B: contiguous over K=H)
//           and bf16 transposed copy Wt[H][V] (for GEMM2 B: contiguous over K=V)
// ---------------------------------------------------------------------------
__global__ __launch_bounds__(256) void convert_w_kernel(
    const float* __restrict__ W, __bf16* __restrict__ Wbf, __bf16* __restrict__ Wt) {
  size_t idx = (size_t)blockIdx.x * 256 + threadIdx.x;
  if (idx >= (size_t)VOC * HD) return;
  size_t v = idx / HD, h = idx - v * HD;
  __bf16 x = (__bf16)W[idx];
  Wbf[idx] = x;
  Wt[h * (size_t)VOC + v] = x;
}

// ---------------------------------------------------------------------------
// Kernel 2: gather embedding rows per token, convert to bf16: Xbf[8192][1024]
// ---------------------------------------------------------------------------
__global__ __launch_bounds__(256) void gather_x_kernel(
    const int* __restrict__ ids, const float* __restrict__ emb, __bf16* __restrict__ Xbf) {
  int tok = blockIdx.x;
  int row = ids[tok];
  const float4* src = (const float4*)(emb + (size_t)row * HD);
  float4 f = src[threadIdx.x];
  __bf16* dst = Xbf + (size_t)tok * HD + (size_t)threadIdx.x * 4;
  dst[0] = (__bf16)f.x; dst[1] = (__bf16)f.y;
  dst[2] = (__bf16)f.z; dst[3] = (__bf16)f.w;
}

// ---------------------------------------------------------------------------
// Kernel 3: fused  logits -> online softmax -> probs @ W  per 16-token M-tile.
//   grid = 512 blocks, block = 256 threads (8 waves).
//   wave w: GEMM1 sub-tile N = vtile + 16w ; GEMM2 output H-chunk [128w,128w+128)
// ---------------------------------------------------------------------------
__global__ __launch_bounds__(256) void mlm_fused_kernel(
    const __bf16* __restrict__ Xbf, const __bf16* __restrict__ Wbf,
    const __bf16* __restrict__ Wt,  const float* __restrict__ bias,
    float* __restrict__ out) {
  __shared__ __bf16 ldsX[16 * HD];       // 32 KB X tile (bf16)
  __shared__ __bf16 ldsP[16 * VTILE];    //  4 KB probs tile (bf16)
  __shared__ float  ldsTmax[NWAVE * 16];
  __shared__ float  ldsTsum[NWAVE * 16];
  __shared__ float  ldsScale[16];
  __shared__ float  rowM[16];
  __shared__ float  rowS[16];

  const int tid   = threadIdx.x;
  const int lane  = tid & 31;
  const int wv    = tid >> 5;     // wave 0..7
  const int nl    = lane & 15;    // column within 16
  const int hi    = lane >> 4;    // half-wave select
  const int mbase = blockIdx.x * 16;

  // ---- stage X tile (16 x 1024 bf16 = 32 KB) into LDS via async DMA -------
  // 2048 x 16B chunks, 8 per thread; tracked with ASYNCcnt (no VGPR round-trip)
  {
    const char* xs = (const char*)(Xbf + (size_t)mbase * HD);
#pragma unroll
    for (int i = 0; i < 8; ++i) {
      const int chunk = tid + i * 256;
      unsigned int      ldsa = (unsigned int)(uintptr_t)ldsX + (unsigned int)chunk * 16u;
      unsigned long long ga  = (unsigned long long)(uintptr_t)(xs) + (unsigned long long)chunk * 16ull;
      asm volatile("global_load_async_to_lds_b128 %0, %1, off"
                   :: "v"(ldsa), "v"(ga) : "memory");
    }
    asm volatile("s_wait_asynccnt 0" ::: "memory");
  }
  if (tid < 16) { rowM[tid] = -__builtin_inff(); rowS[tid] = 0.0f; }
  __syncthreads();

  v8f acc2[8];                            // 16 x 128 fp32 output accumulator
#pragma unroll
  for (int sn = 0; sn < 8; ++sn)
#pragma unroll
    for (int j = 0; j < 8; ++j) acc2[sn][j] = 0.0f;

  for (int vt = 0; vt < VOC / VTILE; ++vt) {
    const int v0 = vt * VTILE;
    const int n0 = v0 + wv * 16;

    // -------- S1: logits sub-tile = X(16xK) * W(n0..n0+15, :)^T, K=1024 ----
    const __bf16* wrow = Wbf + (size_t)(n0 + nl) * HD + hi * 16;
    if (vt + 1 < VOC / VTILE)             // warm L2/WGP$ for next vocab tile
      __builtin_prefetch(Wbf + (size_t)(n0 + VTILE + nl) * HD, 0, 0);
    v8f acc = {0.f, 0.f, 0.f, 0.f, 0.f, 0.f, 0.f, 0.f};
#pragma unroll 4
    for (int k0 = 0; k0 < HD; k0 += 32) {
      v16bf a  = *(const v16bf*)(ldsX + nl * HD + k0 + hi * 16);
      v16bf bm = *(const v16bf*)(wrow + k0);
      acc = __builtin_amdgcn_wmma_f32_16x16x32_bf16(
          false, a, false, bm, (short)0, acc, false, false);
    }
    const float bb = bias[n0 + nl];
#pragma unroll
    for (int j = 0; j < 8; ++j) acc[j] += bb;

    // per-row tile max over 16 columns (a row lives in 16 lanes of one VGPR)
    float rmax[8];
#pragma unroll
    for (int j = 0; j < 8; ++j) {
      float r = acc[j];
      r = fmaxf(r, __shfl_xor(r, 1, 16));
      r = fmaxf(r, __shfl_xor(r, 2, 16));
      r = fmaxf(r, __shfl_xor(r, 4, 16));
      r = fmaxf(r, __shfl_xor(r, 8, 16));
      rmax[j] = r;
    }
    if (nl == 0) {
#pragma unroll
      for (int j = 0; j < 8; ++j) ldsTmax[wv * 16 + j + hi * 8] = rmax[j];
    }
    __syncthreads();

    // -------- S2: cross-wave max, running max update, rescale factor -------
    if (tid < 16) {
      float t = ldsTmax[tid];
#pragma unroll
      for (int w = 1; w < NWAVE; ++w) t = fmaxf(t, ldsTmax[w * 16 + tid]);
      float mo = rowM[tid];
      float mn = fmaxf(mo, t);
      rowM[tid]     = mn;
      ldsScale[tid] = __expf(mo - mn);
    }
    __syncthreads();

    // -------- S3: probs, row sums, rescale output accumulators -------------
    float mn8[8], sc8[8];
#pragma unroll
    for (int j = 0; j < 8; ++j) { mn8[j] = rowM[j + hi * 8]; sc8[j] = ldsScale[j + hi * 8]; }
    float rsum[8];
#pragma unroll
    for (int j = 0; j < 8; ++j) {
      float p = __expf(acc[j] - mn8[j]);
      ldsP[(j + hi * 8) * VTILE + wv * 16 + nl] = (__bf16)p;
      float s = p;
      s += __shfl_xor(s, 1, 16);
      s += __shfl_xor(s, 2, 16);
      s += __shfl_xor(s, 4, 16);
      s += __shfl_xor(s, 8, 16);
      rsum[j] = s;
    }
    if (nl == 0) {
#pragma unroll
      for (int j = 0; j < 8; ++j) ldsTsum[wv * 16 + j + hi * 8] = rsum[j];
    }
#pragma unroll
    for (int sn = 0; sn < 8; ++sn)
#pragma unroll
      for (int j = 0; j < 8; ++j) acc2[sn][j] *= sc8[j];
    __syncthreads();

    // -------- S4: running sum update (wave0) + GEMM2 accumulate ------------
    if (tid < 16) {
      float s = 0.0f;
#pragma unroll
      for (int w = 0; w < NWAVE; ++w) s += ldsTsum[w * 16 + tid];
      rowS[tid] = rowS[tid] * ldsScale[tid] + s;
    }
    // out[16 x 128] += P(16 x 128) * W(v0..v0+127, h-chunk)   via Wt
#pragma unroll
    for (int kk = 0; kk < 4; ++kk) {
      v16bf a2 = *(const v16bf*)(ldsP + nl * VTILE + kk * 32 + hi * 16);
#pragma unroll
      for (int sn = 0; sn < 8; ++sn) {
        const __bf16* wt = Wt + (size_t)(wv * VTILE + sn * 16 + nl) * VOC
                              + v0 + kk * 32 + hi * 16;
        v16bf b2 = *(const v16bf*)wt;
        acc2[sn] = __builtin_amdgcn_wmma_f32_16x16x32_bf16(
            false, a2, false, b2, (short)0, acc2[sn], false, false);
      }
    }
  }
  __syncthreads();

  // -------- finalize: divide by softmax denominator, store fp32 ------------
  float inv8[8];
#pragma unroll
  for (int j = 0; j < 8; ++j) inv8[j] = 1.0f / rowS[j + hi * 8];
#pragma unroll
  for (int sn = 0; sn < 8; ++sn) {
    const int h = wv * VTILE + sn * 16 + nl;
#pragma unroll
    for (int j = 0; j < 8; ++j)
      out[(size_t)(mbase + j + hi * 8) * HD + h] = acc2[sn][j] * inv8[j];
  }
}

// ---------------------------------------------------------------------------
extern "C" void kernel_launch(void* const* d_in, const int* in_sizes, int n_in,
                              void* d_out, int out_size, void* d_ws, size_t ws_size,
                              hipStream_t stream) {
  const int*   ids  = (const int*)d_in[0];      // [4,2048] int32
  const float* emb  = (const float*)d_in[1];    // [32000,1024] fp32
  const float* W    = (const float*)d_in[2];    // [32000,1024] fp32
  const float* bias = (const float*)d_in[3];    // [32000] fp32
  float*       out  = (float*)d_out;            // [4,2048,1024] fp32

  char* ws = (char*)d_ws;
  const size_t wbytes = (size_t)VOC * HD * sizeof(__bf16);   // 65,536,000 B
  __bf16* Wbf = (__bf16*)(ws);
  __bf16* Wt  = (__bf16*)(ws + wbytes);
  __bf16* Xbf = (__bf16*)(ws + 2 * wbytes);                  // + 16.8 MB

  const int convBlocks = (int)(((size_t)VOC * HD + 255) / 256);   // 128000
  convert_w_kernel<<<convBlocks, 256, 0, stream>>>(W, Wbf, Wt);
  gather_x_kernel<<<MTOK, 256, 0, stream>>>(ids, emb, Xbf);
  mlm_fused_kernel<<<MTOK / 16, 256, 0, stream>>>(Xbf, Wbf, Wt, bias, out);
}
